// TransformerBlock_53154515256176
// MI455X (gfx1250) — matrix operand order repaired and checked by reading, not compiled, not run on hardware
//
#include <hip/hip_runtime.h>
#include <hip/hip_bf16.h>

typedef __attribute__((ext_vector_type(16))) _Float16 v16h;
typedef __attribute__((ext_vector_type(8)))  _Float16 v8h;
typedef __attribute__((ext_vector_type(8)))  float    v8f;

union H16 { v16h v; v8h h[2]; };

#define WMMA_F16(A_, B_, C_) \
  __builtin_amdgcn_wmma_f32_16x16x32_f16(false, (A_), false, (B_), (short)0, (C_), false, false)

__device__ __forceinline__ float gelu_approx(float x) {
  const float c = 0.7978845608028654f;  // sqrt(2/pi)
  float u = c * (x + 0.044715f * x * x * x);
  return 0.5f * x * (1.0f + tanhf(u));
}

// ---------------------------------------------------------------------------
// Weight convert + transpose: in[K][N] f32 -> out[N][K] f16  (K-major operand)
// ---------------------------------------------------------------------------
__global__ __launch_bounds__(256) void cvt_transpose_f16(
    const float* __restrict__ in, _Float16* __restrict__ out, int K, int N) {
  size_t idx = (size_t)blockIdx.x * 256u + threadIdx.x;
  size_t k = idx / (unsigned)N;
  size_t n = idx % (unsigned)N;
  out[n * (size_t)K + k] = (_Float16)in[idx];
}

// ---------------------------------------------------------------------------
// LayerNorm over C=768, one block (256 thr) per row, f16 output
// ---------------------------------------------------------------------------
__global__ __launch_bounds__(256) void layernorm_f16_768(
    const float* __restrict__ x, const float* __restrict__ sc,
    const float* __restrict__ sh, _Float16* __restrict__ out) {
  const int C = 768;
  size_t row = blockIdx.x;
  const float* xr = x + row * C;
  float v0 = xr[threadIdx.x];
  float v1 = xr[threadIdx.x + 256];
  float v2 = xr[threadIdx.x + 512];
  float s  = v0 + v1 + v2;
  float s2 = v0 * v0 + v1 * v1 + v2 * v2;
#pragma unroll
  for (int o = 16; o > 0; o >>= 1) {
    s  += __shfl_down(s,  o, 32);
    s2 += __shfl_down(s2, o, 32);
  }
  __shared__ float red[16];
  int wv = threadIdx.x >> 5, ln = threadIdx.x & 31;
  if (ln == 0) { red[wv] = s; red[8 + wv] = s2; }
  __syncthreads();
  s  = red[0]+red[1]+red[2]+red[3]+red[4]+red[5]+red[6]+red[7];
  s2 = red[8]+red[9]+red[10]+red[11]+red[12]+red[13]+red[14]+red[15];
  float mean = s * (1.0f / 768.0f);
  float var  = s2 * (1.0f / 768.0f) - mean * mean;
  float rstd = rsqrtf(var + 1e-5f);
  _Float16* orow = out + row * C;
  orow[threadIdx.x]       = (_Float16)(sc[threadIdx.x]       * (v0 - mean) * rstd + sh[threadIdx.x]);
  orow[threadIdx.x + 256] = (_Float16)(sc[threadIdx.x + 256] * (v1 - mean) * rstd + sh[threadIdx.x + 256]);
  orow[threadIdx.x + 512] = (_Float16)(sc[threadIdx.x + 512] * (v2 - mean) * rstd + sh[threadIdx.x + 512]);
}

// ---------------------------------------------------------------------------
// WMMA GEMM: C[M,N] = A[M,K](f16) x Bt[N,K](f16, pre-transposed), f32 accum.
// Block tile 128x128x32, 256 threads = 8 waves (2x4), wave tile 64x32.
// Software-pipelined (branch-free steady state, last tile peeled).
// ---------------------------------------------------------------------------
enum { EPI_F16 = 0, EPI_BIAS_RES_F32 = 1, EPI_BIAS_GELU_F16 = 2 };

template <int EPI>
__global__ __launch_bounds__(256) void gemm_wmma_f16(
    const _Float16* __restrict__ A, const _Float16* __restrict__ Bt,
    int M, int N, int K,
    _Float16* __restrict__ outh, float* __restrict__ outf,
    const float* __restrict__ bias, const float* __restrict__ res) {
  constexpr int BM = 128, BN = 128, BK = 32, LDP = 40;  // 40-half stride: 20-bank step, conflict-free
  __shared__ _Float16 sA[BM * LDP];
  __shared__ _Float16 sB[BN * LDP];
  const int tid = threadIdx.x;
  const int wave = tid >> 5, lane = tid & 31;
  const int wm = wave >> 2, wn = wave & 3;
  const int l15 = lane & 15, hi = lane >> 4;
  const int m0 = blockIdx.y * BM, n0 = blockIdx.x * BN;

  // This thread's two 8-half chunks within the 128x32 tile
  int rowc[2], kcc[2];
#pragma unroll
  for (int i = 0; i < 2; ++i) {
    int c = tid + i * 256;
    rowc[i] = c >> 2;
    kcc[i]  = (c & 3) * 8;
  }
  const _Float16* gA[2];
  const _Float16* gB[2];
#pragma unroll
  for (int i = 0; i < 2; ++i) {
    gA[i] = A  + (size_t)(m0 + rowc[i]) * K + kcc[i];
    gB[i] = Bt + (size_t)(n0 + rowc[i]) * K + kcc[i];
  }

  const v8f vz = {0.f, 0.f, 0.f, 0.f, 0.f, 0.f, 0.f, 0.f};
  v8f acc[4][2];
#pragma unroll
  for (int i = 0; i < 4; ++i)
#pragma unroll
    for (int j = 0; j < 2; ++j) acc[i][j] = vz;

  // prologue: fetch tile 0, commit to LDS
  v8h ra[2], rb[2];
#pragma unroll
  for (int i = 0; i < 2; ++i) {
    ra[i] = *(const v8h*)(gA[i]);
    rb[i] = *(const v8h*)(gB[i]);
  }
#pragma unroll
  for (int i = 0; i < 2; ++i) {
    *(v8h*)&sA[rowc[i] * LDP + kcc[i]] = ra[i];
    *(v8h*)&sB[rowc[i] * LDP + kcc[i]] = rb[i];
  }
  __syncthreads();

  // steady state: prefetch tile k+1 while computing tile k (branch-free body)
  for (int k0 = 0; k0 < K - BK; k0 += BK) {
#pragma unroll
    for (int i = 0; i < 2; ++i) {
      ra[i] = *(const v8h*)(gA[i] + k0 + BK);
      rb[i] = *(const v8h*)(gB[i] + k0 + BK);
    }
    H16 af[4], bf[2];
#pragma unroll
    for (int i = 0; i < 4; ++i) {  // A frag: K = hi*8 + [0..7], 16+hi*8 + [0..7]
      const _Float16* p = &sA[(wm * 64 + i * 16 + l15) * LDP];
      af[i].h[0] = *(const v8h*)(p + hi * 8);
      af[i].h[1] = *(const v8h*)(p + 16 + hi * 8);
    }
#pragma unroll
    for (int j = 0; j < 2; ++j) {  // B frag: K = hi*8 + [0..7], 16+hi*8 + [0..7]
      const _Float16* p = &sB[(wn * 32 + j * 16 + l15) * LDP];
      bf[j].h[0] = *(const v8h*)(p + hi * 8);
      bf[j].h[1] = *(const v8h*)(p + 16 + hi * 8);
    }
#pragma unroll
    for (int i = 0; i < 4; ++i)
#pragma unroll
      for (int j = 0; j < 2; ++j)
        acc[i][j] = WMMA_F16(af[i].v, bf[j].v, acc[i][j]);
    __syncthreads();  // all waves done reading this tile
#pragma unroll
    for (int i = 0; i < 2; ++i) {
      *(v8h*)&sA[rowc[i] * LDP + kcc[i]] = ra[i];
      *(v8h*)&sB[rowc[i] * LDP + kcc[i]] = rb[i];
    }
    __syncthreads();  // next tile committed for everyone
  }

  // epilogue tile: compute from LDS, no further barriers
  {
    H16 af[4], bf[2];
#pragma unroll
    for (int i = 0; i < 4; ++i) {
      const _Float16* p = &sA[(wm * 64 + i * 16 + l15) * LDP];
      af[i].h[0] = *(const v8h*)(p + hi * 8);
      af[i].h[1] = *(const v8h*)(p + 16 + hi * 8);
    }
#pragma unroll
    for (int j = 0; j < 2; ++j) {
      const _Float16* p = &sB[(wn * 32 + j * 16 + l15) * LDP];
      bf[j].h[0] = *(const v8h*)(p + hi * 8);
      bf[j].h[1] = *(const v8h*)(p + 16 + hi * 8);
    }
#pragma unroll
    for (int i = 0; i < 4; ++i)
#pragma unroll
      for (int j = 0; j < 2; ++j)
        acc[i][j] = WMMA_F16(af[i].v, bf[j].v, acc[i][j]);
  }

#pragma unroll
  for (int i = 0; i < 4; ++i) {
#pragma unroll
    for (int j = 0; j < 2; ++j) {
      int n = n0 + wn * 32 + j * 16 + l15;
      float bv = (EPI == EPI_F16) ? 0.f : bias[n];
#pragma unroll
      for (int r = 0; r < 8; ++r) {
        int m = m0 + wm * 64 + i * 16 + r + 8 * hi;
        size_t off = (size_t)m * N + n;
        float val = acc[i][j][r];
        if (EPI == EPI_F16)                outh[off] = (_Float16)val;
        else if (EPI == EPI_BIAS_RES_F32)  outf[off] = val + bv + res[off];
        else                               outh[off] = (_Float16)gelu_approx(val + bv);
      }
    }
  }
}

// ---------------------------------------------------------------------------
// Flash attention: block = 4 waves, one (b, head, 64-row Q tile) per block.
// Each wave owns a 16-row Q strip; K tile + V^T tile staged in LDS.
// ---------------------------------------------------------------------------
__global__ __launch_bounds__(128) void attn_flash(
    const _Float16* __restrict__ q, const _Float16* __restrict__ kk,
    const _Float16* __restrict__ vv, _Float16* __restrict__ ctx) {
  constexpr int T = 1024, E = 768, HDIM = 64, NH = 12, QB = 64, LDK = 72;
  __shared__ _Float16 sK[64 * LDK];
  __shared__ _Float16 sVT[64 * LDK];
  __shared__ _Float16 sP[4 * 16 * LDK];

  const int QT = T / QB;  // 16
  int qt = blockIdx.x % QT;
  int bh = blockIdx.x / QT;
  int b = bh / NH, h = bh % NH;
  const size_t headoff = (size_t)b * T * E + (size_t)h * HDIM;
  const int tid = threadIdx.x, wave = tid >> 5, lane = tid & 31;
  const int l15 = lane & 15, hi = lane >> 4;
  const int q0 = qt * QB + wave * 16;

  H16 aq[2];
  {
    const _Float16* qrow = q + headoff + (size_t)(q0 + l15) * E;
#pragma unroll
    for (int ks = 0; ks < 2; ++ks) {
      aq[ks].h[0] = *(const v8h*)(qrow + ks * 32 + hi * 8);
      aq[ks].h[1] = *(const v8h*)(qrow + ks * 32 + 16 + hi * 8);
    }
  }
  const v8f vz = {0.f, 0.f, 0.f, 0.f, 0.f, 0.f, 0.f, 0.f};
  float m_i[8], l_i[8];
  v8f O[4];
#pragma unroll
  for (int r = 0; r < 8; ++r) { m_i[r] = -3.0e38f; l_i[r] = 0.f; }
#pragma unroll
  for (int t = 0; t < 4; ++t) O[t] = vz;

  for (int kb = 0; kb <= qt; ++kb) {
    {  // cooperative stage: K -> sK[key][d], V -> sVT[d][key]
      int row = tid >> 1;
      int dc = (tid & 1) * 32;
      const _Float16* kr = kk + headoff + (size_t)(kb * QB + row) * E + dc;
#pragma unroll
      for (int c = 0; c < 4; ++c)
        *(v8h*)&sK[row * LDK + dc + c * 8] = *(const v8h*)(kr + c * 8);
      const _Float16* vr = vv + headoff + (size_t)(kb * QB + row) * E + dc;
      v8h t4[4];
#pragma unroll
      for (int c = 0; c < 4; ++c) t4[c] = *(const v8h*)(vr + c * 8);
#pragma unroll
      for (int c = 0; c < 4; ++c)
#pragma unroll
        for (int e = 0; e < 8; ++e)
          sVT[(dc + c * 8 + e) * LDK + row] = t4[c][e];
    }
    __syncthreads();

    v8f S[4];
#pragma unroll
    for (int j = 0; j < 4; ++j) {  // S = Q . K^T  (B operand from sK rows)
      const _Float16* kp = &sK[(j * 16 + l15) * LDK];
      H16 b0, b1;
      b0.h[0] = *(const v8h*)(kp + hi * 8);
      b0.h[1] = *(const v8h*)(kp + 16 + hi * 8);
      b1.h[0] = *(const v8h*)(kp + 32 + hi * 8);
      b1.h[1] = *(const v8h*)(kp + 32 + 16 + hi * 8);
      v8f z = vz;
      z = WMMA_F16(aq[0].v, b0.v, z);
      z = WMMA_F16(aq[1].v, b1.v, z);
      S[j] = z;
    }
    const float scl = 0.125f;  // 1/sqrt(64)
    const bool diag = (kb == qt);
#pragma unroll
    for (int j = 0; j < 4; ++j)
#pragma unroll
      for (int r = 0; r < 8; ++r) {
        float x = S[j][r] * scl;
        if (diag) {
          int qr = q0 + r + 8 * hi;
          int kc = kb * QB + j * 16 + l15;
          if (kc > qr) x = -1.0e30f;
        }
        S[j][r] = x;
      }
    // online softmax; lanes 0-15 hold row r, lanes 16-31 hold row r+8:
    // shfl_xor masks 1,2,4,8 reduce within each 16-lane half.
#pragma unroll
    for (int r = 0; r < 8; ++r) {
      float mx = fmaxf(fmaxf(S[0][r], S[1][r]), fmaxf(S[2][r], S[3][r]));
#pragma unroll
      for (int o = 1; o < 16; o <<= 1) mx = fmaxf(mx, __shfl_xor(mx, o, 32));
      float nm = fmaxf(m_i[r], mx);
      float corr = __expf(m_i[r] - nm);
      m_i[r] = nm;
      float rs = 0.f;
#pragma unroll
      for (int j = 0; j < 4; ++j) {
        float p = __expf(S[j][r] - nm);
        S[j][r] = p;
        rs += p;
      }
#pragma unroll
      for (int o = 1; o < 16; o <<= 1) rs += __shfl_xor(rs, o, 32);
      l_i[r] = l_i[r] * corr + rs;
#pragma unroll
      for (int t = 0; t < 4; ++t) O[t][r] *= corr;
    }
    // P (C layout) -> per-wave LDS -> reload in A-fragment layout
    _Float16* pw = &sP[wave * 16 * LDK];
#pragma unroll
    for (int j = 0; j < 4; ++j)
#pragma unroll
      for (int r = 0; r < 8; ++r)
        pw[(r + 8 * hi) * LDK + j * 16 + l15] = (_Float16)S[j][r];
    H16 ap[2];
    {
      const _Float16* pr = pw + (size_t)l15 * LDK;
#pragma unroll
      for (int ks = 0; ks < 2; ++ks) {
        ap[ks].h[0] = *(const v8h*)(pr + ks * 32 + hi * 8);
        ap[ks].h[1] = *(const v8h*)(pr + ks * 32 + 16 + hi * 8);
      }
    }
#pragma unroll
    for (int t = 0; t < 4; ++t) {  // O += P . V  (B operand from sVT rows)
      const _Float16* vp = &sVT[(t * 16 + l15) * LDK];
      H16 b0, b1;
      b0.h[0] = *(const v8h*)(vp + hi * 8);
      b0.h[1] = *(const v8h*)(vp + 16 + hi * 8);
      b1.h[0] = *(const v8h*)(vp + 32 + hi * 8);
      b1.h[1] = *(const v8h*)(vp + 32 + 16 + hi * 8);
      O[t] = WMMA_F16(ap[0].v, b0.v, O[t]);
      O[t] = WMMA_F16(ap[1].v, b1.v, O[t]);
    }
    __syncthreads();
  }
#pragma unroll
  for (int t = 0; t < 4; ++t)
#pragma unroll
    for (int r = 0; r < 8; ++r) {
      float o = O[t][r] / l_i[r];
      ctx[headoff + (size_t)(q0 + r + 8 * hi) * E + t * 16 + l15] = (_Float16)o;
    }
}

// ---------------------------------------------------------------------------
extern "C" void kernel_launch(void* const* d_in, const int* in_sizes, int n_in,
                              void* d_out, int out_size, void* d_ws, size_t ws_size,
                              hipStream_t stream) {
  (void)in_sizes; (void)n_in; (void)out_size; (void)ws_size;
  const float* x  = (const float*)d_in[0];
  const float* Wq = (const float*)d_in[1];
  const float* Wk = (const float*)d_in[2];
  const float* Wv = (const float*)d_in[3];
  const float* Wo = (const float*)d_in[4];
  const float* bo = (const float*)d_in[5];
  const float* W1 = (const float*)d_in[6];
  const float* b1 = (const float*)d_in[7];
  const float* W2 = (const float*)d_in[8];
  const float* b2 = (const float*)d_in[9];
  const float* g1 = (const float*)d_in[10];
  const float* s1 = (const float*)d_in[11];
  const float* g2 = (const float*)d_in[12];
  const float* s2 = (const float*)d_in[13];

  const int Bb = 8, T = 1024, E = 768, FF = 3072, NH = 12;
  const int M = Bb * T;  // 8192

  char* ws = (char*)d_ws;
  size_t off = 0;
  auto take = [&](size_t bytes) -> char* {
    char* p = ws + off;
    off = (off + bytes + 255) & ~(size_t)255;
    return p;
  };
  _Float16* wq_t = (_Float16*)take((size_t)E * E * 2);
  _Float16* wk_t = (_Float16*)take((size_t)E * E * 2);
  _Float16* wv_t = (_Float16*)take((size_t)E * E * 2);
  _Float16* wo_t = (_Float16*)take((size_t)E * E * 2);
  _Float16* w1_t = (_Float16*)take((size_t)E * FF * 2);
  _Float16* w2_t = (_Float16*)take((size_t)FF * E * 2);
  _Float16* h1   = (_Float16*)take((size_t)M * E * 2);
  _Float16* qf   = (_Float16*)take((size_t)M * E * 2);
  _Float16* kf   = (_Float16*)take((size_t)M * E * 2);
  _Float16* vf   = (_Float16*)take((size_t)M * E * 2);
  _Float16* ctxh = (_Float16*)take((size_t)M * E * 2);
  float*    x2   = (float*)   take((size_t)M * E * 4);
  _Float16* h2   = (_Float16*)take((size_t)M * E * 2);
  _Float16* ff1  = (_Float16*)take((size_t)M * FF * 2);

  // Weight convert + transpose to f16 K-major
  cvt_transpose_f16<<<(E * E) / 256, 256, 0, stream>>>(Wq, wq_t, E, E);
  cvt_transpose_f16<<<(E * E) / 256, 256, 0, stream>>>(Wk, wk_t, E, E);
  cvt_transpose_f16<<<(E * E) / 256, 256, 0, stream>>>(Wv, wv_t, E, E);
  cvt_transpose_f16<<<(E * E) / 256, 256, 0, stream>>>(Wo, wo_t, E, E);
  cvt_transpose_f16<<<(E * FF) / 256, 256, 0, stream>>>(W1, w1_t, E, FF);
  cvt_transpose_f16<<<(FF * E) / 256, 256, 0, stream>>>(W2, w2_t, FF, E);

  dim3 g768(E / 128, M / 128);    // (6, 64)
  dim3 g3072(FF / 128, M / 128);  // (24, 64)

  // LN1 -> QKV projections
  layernorm_f16_768<<<M, 256, 0, stream>>>(x, g1, s1, h1);
  gemm_wmma_f16<EPI_F16><<<g768, 256, 0, stream>>>(h1, wq_t, M, E, E, qf, nullptr, nullptr, nullptr);
  gemm_wmma_f16<EPI_F16><<<g768, 256, 0, stream>>>(h1, wk_t, M, E, E, kf, nullptr, nullptr, nullptr);
  gemm_wmma_f16<EPI_F16><<<g768, 256, 0, stream>>>(h1, wv_t, M, E, E, vf, nullptr, nullptr, nullptr);

  // Causal flash attention
  attn_flash<<<Bb * NH * (T / 64), 128, 0, stream>>>(qf, kf, vf, ctxh);

  // Output projection + residual, LN2, FFN
  gemm_wmma_f16<EPI_BIAS_RES_F32><<<g768, 256, 0, stream>>>(ctxh, wo_t, M, E, E, nullptr, x2, bo, x);
  layernorm_f16_768<<<M, 256, 0, stream>>>(x2, g2, s2, h2);
  gemm_wmma_f16<EPI_BIAS_GELU_F16><<<g3072, 256, 0, stream>>>(h2, w1_t, M, FF, E, ff1, nullptr, b1, nullptr);
  gemm_wmma_f16<EPI_BIAS_RES_F32><<<g768, 256, 0, stream>>>(ff1, w2_t, M, E, FF, nullptr, (float*)d_out, b2, x2);
}
